// Attention_22943715295808
// MI455X (gfx1250) — matrix operand order repaired and checked by reading, not compiled, not run on hardware
//
#include <hip/hip_runtime.h>

// ---------------------------------------------------------------------------
// MHA forward for MI455X (gfx1250, wave32).
//   q = X Wq + b -> [B*N][S][H] bf16   (A/B fragments of S = Q K^T)
//   k = X Wk + b -> [B*N][S][H] bf16
//   v = X Wv + b -> [B*N][H][S] bf16   (transposed: B fragments of O = P V)
//   flash-attention: 32 query rows per wave (2 M-tiles) so each K/V fragment
//   feeds 2 WMMAs; scores never touch HBM.
// All GEMMs via v_wmma_f32_16x16x32_bf16 (f32 accumulate).
// ---------------------------------------------------------------------------

typedef __attribute__((ext_vector_type(16))) __bf16 v16bf;
typedef __attribute__((ext_vector_type(2)))  __bf16 v2bf;
typedef __attribute__((ext_vector_type(8)))  float  v8f;

#define BATCH  4
#define SEQ    2048
#define NHEADS 12
#define HDIM   64
#define DDIM   768   // NHEADS * HDIM

__device__ __forceinline__ v8f wmma_bf16(v16bf a, v16bf b, v8f c) {
  // (neg_a, A, neg_b, B, c_mod, C, reuse_a, reuse_b)
  return __builtin_amdgcn_wmma_f32_16x16x32_bf16(false, a, false, b,
                                                 (short)0, c, false, false);
}

// Hardware K-position for 16-bit A fragment (16x32), per ISA 7.12.2:
// lanes 0-15: V0-3 -> K 0..7,  V4-7 -> K 16..23
// lanes16-31: V0-3 -> K 8..15, V4-7 -> K 24..31
__device__ __forceinline__ int a_kpos(int v, int hi) {
  return ((v & 4) << 2) + hi * 8 + ((v & 3) << 1);
}

// Load A fragment (M x K = 16 x 32) from row-major bf16 [row][k], row stride ld.
__device__ __forceinline__ v16bf load_a_frag(const __bf16* __restrict__ src, int ld) {
  const int lane = threadIdx.x & 31;
  const int m = lane & 15, hi = lane >> 4;
  const __bf16* row = src + m * ld;
  v16bf a;
#pragma unroll
  for (int v = 0; v < 8; ++v) {
    const int k = hi * 16 + 2 * v;
    a[2 * v]     = row[k];
    a[2 * v + 1] = row[k + 1];
  }
  return a;
}

// Load B fragment (K x N = 32 x 16) from an "N-major" source srcT[n][kk]
// (srcT[n][kk] == B[kk][n]), n stride = ld.
// Contiguous k order (as load_a_frag): lane (hi*16+n), VGPR v holds B[hi*16+2v][n], B[hi*16+2v+1][n].
__device__ __forceinline__ v16bf load_b_frag(const __bf16* __restrict__ srcT, int ld) {
  const int lane = threadIdx.x & 31;
  const int n = lane & 15, hi = lane >> 4;
  const __bf16* col = srcT + n * ld + hi * 16;
  v16bf b;
#pragma unroll
  for (int v = 0; v < 8; ++v) {
    b[2 * v]     = col[2 * v];
    b[2 * v + 1] = col[2 * v + 1];
  }
  return b;
}

// ---------------------------------------------------------------------------
// Projection GEMM: out = X[8192x768] @ W[768x768] + bias, stored bf16.
// mode 0: out[((b*N+n)*SEQ + s)*H + h]   (Q, K layout)
// mode 1: out[((b*N+n)*H + h)*SEQ + s]   (V layout, transposed)
// Block: 128 threads (4 waves), 64x64 tile; wave quadrant 32x32 = 2x2 WMMA.
// LDS staging uses packed 2x bf16 stores (one b32 per pair).
// ---------------------------------------------------------------------------
__global__ __launch_bounds__(128)
void proj_kernel(const float* __restrict__ X, const float* __restrict__ W,
                 const float* __restrict__ bias, __bf16* __restrict__ out,
                 int transposed) {
  __shared__ __bf16 lds_x[64][36];   // X tile, row-major [m][k]
  __shared__ __bf16 lds_w[64][36];   // W tile transposed: lds_w[n][k] = W[k0+k][n0+n]

  const int m0 = blockIdx.x * 64;
  const int n0 = blockIdx.y * 64;
  const int tid  = threadIdx.x;
  const int wave = tid >> 5;
  const int lane = tid & 31;
  const int wm = (wave >> 1) * 32;   // quadrant row offset within tile
  const int wn = (wave & 1) * 32;    // quadrant col offset

  v8f acc[2][2] = {};

  for (int k0 = 0; k0 < DDIM; k0 += 32) {
    // Stage X tile (64x32 f32 -> bf16), float2 loads, packed b32 LDS stores.
#pragma unroll
    for (int i = 0; i < 8; ++i) {
      const int idx = tid + i * 128;        // pair index 0..1023
      const int r = idx >> 4;               // 16 pairs per 32-wide row
      const int c = (idx & 15) * 2;
      const float2 xv = *(const float2*)&X[(size_t)(m0 + r) * DDIM + (k0 + c)];
      v2bf p; p[0] = (__bf16)xv.x; p[1] = (__bf16)xv.y;
      *(v2bf*)&lds_x[r][c] = p;
    }
    // Stage W tile transposed: pack along k so stores are b32; reads coalesce
    // over n (consecutive tids -> consecutive columns of W).
#pragma unroll
    for (int i = 0; i < 8; ++i) {
      const int idx = tid + i * 128;        // 0..1023
      const int kp = idx >> 6;              // k pair 0..15
      const int n  = idx & 63;
      const float w0 = W[(size_t)(k0 + 2 * kp)     * DDIM + (n0 + n)];
      const float w1 = W[(size_t)(k0 + 2 * kp + 1) * DDIM + (n0 + n)];
      v2bf p; p[0] = (__bf16)w0; p[1] = (__bf16)w1;
      *(v2bf*)&lds_w[n][2 * kp] = p;
    }
    __syncthreads();

#pragma unroll
    for (int mi = 0; mi < 2; ++mi) {
      const v16bf a = load_a_frag(&lds_x[wm + mi * 16][0], 36);
#pragma unroll
      for (int ni = 0; ni < 2; ++ni) {
        const v16bf b = load_b_frag(&lds_w[wn + ni * 16][0], 36);
        acc[mi][ni] = wmma_bf16(a, b, acc[mi][ni]);
      }
    }
    __syncthreads();
  }

  // Store with bias. C layout: elem r -> row r + (lane>=16)*8, col = lane&15.
  const int rr   = (lane >> 4) * 8;
  const int ncol = lane & 15;
#pragma unroll
  for (int mi = 0; mi < 2; ++mi) {
#pragma unroll
    for (int ni = 0; ni < 2; ++ni) {
#pragma unroll
      for (int r = 0; r < 8; ++r) {
        const int m   = m0 + wm + mi * 16 + rr + r;     // global row (b*SEQ + s)
        const int col = n0 + wn + ni * 16 + ncol;       // global col (n*H + h)
        const float val = acc[mi][ni][r] + bias[col];
        const int b = m >> 11, s = m & (SEQ - 1);
        const int n = col >> 6, h = col & (HDIM - 1);
        const size_t addr = transposed
            ? (((size_t)(b * NHEADS + n) * HDIM + h) * SEQ + s)
            : (((size_t)(b * NHEADS + n) * SEQ + s) * HDIM + h);
        out[addr] = (__bf16)val;
      }
    }
  }
}

// ---------------------------------------------------------------------------
// Flash attention: grid (B*N, SEQ/128); 128 threads = 4 waves x 32 query rows.
// Per 32-key tile: 4 K-fragments feed 8 S-WMMAs, 4 V-fragments feed 8 PV-WMMAs
// (each streamed fragment reused across the wave's 2 M-tiles).
// ---------------------------------------------------------------------------
__global__ __launch_bounds__(128)
void attn_kernel(const __bf16* __restrict__ Qp, const __bf16* __restrict__ Kp,
                 const __bf16* __restrict__ Vp, const float* __restrict__ mask,
                 float* __restrict__ out) {
  __shared__ __bf16 lds_p[4][2][16][36];   // per-wave, per-Mtile P re-swizzle tile

  const int bn = blockIdx.x;                 // b*NHEADS + n
  const int b  = bn / NHEADS;
  const int n  = bn - b * NHEADS;
  const int wave = threadIdx.x >> 5;
  const int lane = threadIdx.x & 31;
  const int q0 = blockIdx.y * 128 + wave * 32;    // first query row of this wave
  const int rr   = (lane >> 4) * 8;               // C-layout row offset for this lane
  const int ncol = lane & 15;                     // C-layout column

  const __bf16* qbase = Qp + ((size_t)bn * SEQ + q0) * HDIM;
  const __bf16* kbase = Kp + (size_t)bn * SEQ * HDIM;
  const __bf16* vbase = Vp + (size_t)bn * HDIM * SEQ;

  // Q fragments: [mtile][h-chunk], resident for the whole K loop.
  v16bf qa[2][2];
#pragma unroll
  for (int mt = 0; mt < 2; ++mt) {
    qa[mt][0] = load_a_frag(qbase + (size_t)mt * 16 * HDIM, HDIM);
    qa[mt][1] = load_a_frag(qbase + (size_t)mt * 16 * HDIM + 32, HDIM);
  }

  float m_i[2][8], l_i[2][8];
#pragma unroll
  for (int mt = 0; mt < 2; ++mt)
#pragma unroll
    for (int r = 0; r < 8; ++r) { m_i[mt][r] = -1.0e30f; l_i[mt][r] = 0.0f; }
  v8f o[2][4] = {};

  for (int k0 = 0; k0 < SEQ; k0 += 32) {
    // ---- S = (Q K^T): K fragments shared across both M-tiles --------------
    v8f s[2][2] = {};   // [mtile][key-halftile]
#pragma unroll
    for (int t = 0; t < 2; ++t) {
#pragma unroll
      for (int hc = 0; hc < 2; ++hc) {
        const v16bf kb =
            load_b_frag(kbase + (size_t)(k0 + t * 16) * HDIM + hc * 32, HDIM);
#pragma unroll
        for (int mt = 0; mt < 2; ++mt)
          s[mt][t] = wmma_bf16(qa[mt][hc], kb, s[mt][t]);
      }
    }

    // ---- online softmax per M-tile (rows split across lane halves) --------
#pragma unroll
    for (int mt = 0; mt < 2; ++mt) {
#pragma unroll
      for (int r = 0; r < 8; ++r) {
        const int qrow = q0 + mt * 16 + rr + r;
        const float* mrow = mask + ((size_t)b * SEQ + qrow) * SEQ + k0;
        float s0 = s[mt][0][r] * 0.125f + mrow[ncol];        // 1/sqrt(64)
        float s1 = s[mt][1][r] * 0.125f + mrow[16 + ncol];

        float mx = fmaxf(s0, s1);
#pragma unroll
        for (int off = 1; off < 16; off <<= 1)
          mx = fmaxf(mx, __shfl_xor(mx, off, 32));
        const float mnew = fmaxf(m_i[mt][r], mx);
        const float corr = __expf(m_i[mt][r] - mnew);
        const float p0 = __expf(s0 - mnew);
        const float p1 = __expf(s1 - mnew);
        float rs = p0 + p1;
#pragma unroll
        for (int off = 1; off < 16; off <<= 1)
          rs += __shfl_xor(rs, off, 32);
        l_i[mt][r] = l_i[mt][r] * corr + rs;
        m_i[mt][r] = mnew;
#pragma unroll
        for (int ht = 0; ht < 4; ++ht) o[mt][ht][r] *= corr;

        // scatter P (bf16) into per-wave LDS tile in C layout positions
        lds_p[wave][mt][rr + r][ncol]      = (__bf16)p0;
        lds_p[wave][mt][rr + r][16 + ncol] = (__bf16)p1;
      }
    }

    // ---- O += P * V: V fragments shared across both M-tiles ---------------
    v16bf pa[2];
    pa[0] = load_a_frag(&lds_p[wave][0][0][0], 36);
    pa[1] = load_a_frag(&lds_p[wave][1][0][0], 36);
#pragma unroll
    for (int ht = 0; ht < 4; ++ht) {
      const v16bf vb = load_b_frag(vbase + (size_t)(ht * 16) * SEQ + k0, SEQ);
#pragma unroll
      for (int mt = 0; mt < 2; ++mt)
        o[mt][ht] = wmma_bf16(pa[mt], vb, o[mt][ht]);
    }
  }

  // ---- normalize + store: out[b*SEQ + q][n*H + h] -------------------------
#pragma unroll
  for (int mt = 0; mt < 2; ++mt) {
#pragma unroll
    for (int r = 0; r < 8; ++r) {
      const float inv = 1.0f / l_i[mt][r];
      const int qrow = q0 + mt * 16 + rr + r;
      float* orow = out + (size_t)(b * SEQ + qrow) * DDIM + n * HDIM;
#pragma unroll
      for (int ht = 0; ht < 4; ++ht)
        orow[ht * 16 + ncol] = o[mt][ht][r] * inv;
    }
  }
}

// ---------------------------------------------------------------------------
extern "C" void kernel_launch(void* const* d_in, const int* in_sizes, int n_in,
                              void* d_out, int out_size, void* d_ws, size_t ws_size,
                              hipStream_t stream) {
  (void)in_sizes; (void)n_in; (void)out_size; (void)ws_size;
  const float* query = (const float*)d_in[0];
  const float* key_t = (const float*)d_in[1];
  const float* value = (const float*)d_in[2];
  const float* mask  = (const float*)d_in[3];
  const float* Wq = (const float*)d_in[4];
  const float* bq = (const float*)d_in[5];
  const float* Wk = (const float*)d_in[6];
  const float* bk = (const float*)d_in[7];
  const float* Wv = (const float*)d_in[8];
  const float* bv = (const float*)d_in[9];
  float* outp = (float*)d_out;

  const size_t tsz = (size_t)BATCH * NHEADS * SEQ * HDIM;   // 6.29M bf16 each
  __bf16* qp = (__bf16*)d_ws;
  __bf16* kp = qp + tsz;
  __bf16* vp = kp + tsz;

  dim3 pgrid((BATCH * SEQ) / 64, DDIM / 64);   // (128, 12)
  proj_kernel<<<pgrid, 128, 0, stream>>>(query, Wq, bq, qp, 0);
  proj_kernel<<<pgrid, 128, 0, stream>>>(key_t, Wk, bk, kp, 0);
  proj_kernel<<<pgrid, 128, 0, stream>>>(value, Wv, bv, vp, 1);

  dim3 agrid(BATCH * NHEADS, SEQ / 128);       // (48, 16)
  attn_kernel<<<agrid, 128, 0, stream>>>(qp, kp, vp, mask, outp);
}
